// SocialLSTM_28381143892375
// MI455X (gfx1250) — matrix operand order repaired and checked by reading, not compiled, not run on hardware
//
#include <hip/hip_runtime.h>
#include <math.h>

// ---------------- types for CDNA5 WMMA ----------------
typedef __bf16 bf16_t;
typedef __attribute__((ext_vector_type(16))) __bf16 v16bf;
typedef __attribute__((ext_vector_type(8)))  float  v8f;
typedef __attribute__((ext_vector_type(4)))  unsigned int u32x4;

union AB16 { v16bf v; u32x4 u[2]; };

__device__ __forceinline__ v16bf load_a_frag(const bf16_t* p0, const bf16_t* p1) {
  AB16 t;
  t.u[0] = *(const u32x4*)p0;   // K = k0 + 16*half .. +7   (row M = lane&15)
  t.u[1] = *(const u32x4*)p1;   // K = k0 + 16*half + 8 .. +7
  return t.v;
}
__device__ __forceinline__ v16bf load_b_frag(const bf16_t* p) {
  AB16 t;                        // 16 contiguous bf16 of weight row n = lane&15
  t.u[0] = *(const u32x4*)p;
  t.u[1] = *(const u32x4*)(p + 8);
  return t.v;
}
__device__ __forceinline__ v8f wmma_bf16(v16bf a, v16bf b, v8f c) {
  return __builtin_amdgcn_wmma_f32_16x16x32_bf16(false, a, false, b, (short)0, c, false, false);
}
__device__ __forceinline__ v8f vzero8() {
  v8f z = {0.f,0.f,0.f,0.f,0.f,0.f,0.f,0.f};
  return z;
}
__device__ __forceinline__ float sigm(float x) { return 1.0f / (1.0f + expf(-x)); }

// ---------------- problem constants ----------------
#define NAG      1024
#define TOBS     8
#define PREDLEN  12
#define HID      128
#define EMB      64
#define XW       256      // [et(64)|at(64)|h(128)]
#define ZW       512      // 4*HID
#define POOLW    512      // GRID*GRID*HID
#define SCN      64       // agents per scene

// =====================================================================
// prep: bf16 weight staging + h/c broadcast init
//   Wz[512][256] = [Wih | Whh] (row-major, bf16)
//   socWb[64][512] = soc_W (bf16)
// launch: 512 blocks x 256 (131072 threads)
// =====================================================================
__global__ void sl_prep(const float* __restrict__ Wih, const float* __restrict__ Whh,
                        const float* __restrict__ socW,
                        const float* __restrict__ h0, const float* __restrict__ c0,
                        bf16_t* __restrict__ Wz, bf16_t* __restrict__ socWb,
                        float* __restrict__ ht, float* __restrict__ ct) {
  int idx = blockIdx.x * blockDim.x + threadIdx.x;
  if (idx < ZW * XW) {
    int j = idx >> 8, k = idx & 255;
    float v = (k < HID) ? Wih[j * HID + k] : Whh[j * HID + (k - HID)];
    Wz[idx] = (bf16_t)v;
  }
  if (idx < EMB * POOLW) socWb[idx] = (bf16_t)socW[idx];
  if (idx < NAG * HID) { ht[idx] = h0[idx & (HID - 1)]; ct[idx] = c0[idx & (HID - 1)]; }
}

// =====================================================================
// social pooling: one block per scene (16 blocks x 256 threads)
// pooled[i][g*128+hd] (bf16) = sum_j pair(i,j,g) * ht[j][hd]
// =====================================================================
__global__ void sl_pool(const float* __restrict__ ht,
                        const float* __restrict__ pos_base, int pos_stride,
                        const int* __restrict__ mask_base, int mask_stride,
                        const int* __restrict__ scene,
                        bf16_t* __restrict__ pooled) {
  __shared__ float sh_ht[SCN * HID];
  __shared__ float sh_px[SCN], sh_py[SCN];
  __shared__ int   sh_sc[SCN], sh_act[SCN];
  __shared__ unsigned char sh_pair[SCN * SCN];

  int tid  = threadIdx.x;
  int base = blockIdx.x * SCN;

  for (int idx = tid; idx < SCN * HID; idx += 256) sh_ht[idx] = ht[base * HID + idx];
  if (tid < SCN) {
    int row = base + tid;
    sh_px[tid]  = pos_base[row * pos_stride + 0];
    sh_py[tid]  = pos_base[row * pos_stride + 1];
    sh_sc[tid]  = scene[row];
    sh_act[tid] = mask_base[row * mask_stride];
  }
  __syncthreads();

  const float halfspan = 1.0f, eps = 0.01f;   // SPAN=2, cell=1
  for (int pid = tid; pid < SCN * SCN; pid += 256) {
    int i = pid >> 6, j = pid & 63;
    float rx = sh_px[j] - sh_px[i];
    float ry = sh_py[j] - sh_py[i];
    bool within = (rx < halfspan - eps) && (rx > -halfspan + eps) &&
                  (ry < halfspan - eps) && (ry > -halfspan + eps);
    bool ok = within && (i != j) && (sh_sc[i] == sh_sc[j]) &&
              (sh_act[i] == 1) && (sh_act[j] == 1);
    int gx = (int)floorf(rx + halfspan);      // /cell with cell==1
    int gy = (int)floorf(ry + halfspan);
    int gid = gx * 2 + gy;
    gid = gid < 0 ? 0 : (gid > 3 ? 3 : gid);
    sh_pair[pid] = ok ? (unsigned char)gid : (unsigned char)255;
  }
  __syncthreads();

  for (int widx = tid; widx < SCN * HID; widx += 256) {
    int i = widx >> 7, hd = widx & (HID - 1);
    float a0 = 0.f, a1 = 0.f, a2 = 0.f, a3 = 0.f;
    const unsigned char* pr = &sh_pair[i * SCN];
    for (int j = 0; j < SCN; ++j) {
      unsigned char g = pr[j];
      if (g < 4) {
        float v = sh_ht[j * HID + hd];
        if      (g == 0) a0 += v;
        else if (g == 1) a1 += v;
        else if (g == 2) a2 += v;
        else             a3 += v;
      }
    }
    size_t o = (size_t)(base + i) * POOLW + hd;
    pooled[o +   0] = (bf16_t)a0;
    pooled[o + 128] = (bf16_t)a1;
    pooled[o + 256] = (bf16_t)a2;
    pooled[o + 384] = (bf16_t)a3;
  }
}

// =====================================================================
// fused step: et + at(WMMA) -> xh in LDS -> z(WMMA) -> gates -> h/c
// 1 wave owns 16 agents. block = 128 (4 waves), grid = 16 -> 1024 rows.
// =====================================================================
__global__ void __launch_bounds__(128)
sl_step(const float* __restrict__ rel_base, int rel_stride,
        const int* __restrict__ mask_base, int mask_stride,
        const bf16_t* __restrict__ pooled,
        const bf16_t* __restrict__ Wz, const bf16_t* __restrict__ socWb,
        const float* __restrict__ posW, const float* __restrict__ posb,
        const float* __restrict__ socb,
        const float* __restrict__ bih, const float* __restrict__ bhh,
        float* __restrict__ ht, float* __restrict__ ct) {
  __shared__ bf16_t xh[4 * 16 * XW];          // 32 KB

  int tid  = threadIdx.x;
  int lane = tid & 31;
  int w    = tid >> 5;
  int rowBase = (blockIdx.x * 4 + w) * 16;
  int m16  = lane & 15;
  int half = lane >> 4;
  bf16_t* xrow = &xh[w * 16 * XW];

  // ---- et = relu(rel @ posW^T + posb) -> cols [0,64) ----
  for (int idx = lane; idx < 16 * EMB; idx += 32) {
    int r = idx >> 6, j = idx & 63;
    int row = rowBase + r;
    float r0 = rel_base[row * rel_stride + 0];
    float r1 = rel_base[row * rel_stride + 1];
    float e = r0 * posW[j * 2 + 0] + r1 * posW[j * 2 + 1] + posb[j];
    xrow[r * XW + j] = (bf16_t)fmaxf(e, 0.0f);
  }
  // ---- h -> cols [128,256) ----
  for (int idx = lane; idx < 16 * HID; idx += 32) {
    int r = idx >> 7, j = idx & (HID - 1);
    xrow[r * XW + 128 + j] = (bf16_t)ht[(rowBase + r) * HID + j];
  }

  // ---- at = relu(pooled @ socW^T + socb) via WMMA, K=512 -> cols [64,128) ----
  {
    v8f acc0 = vzero8(), acc1 = vzero8(), acc2 = vzero8(), acc3 = vzero8();
    const bf16_t* arow = pooled + (size_t)(rowBase + m16) * POOLW;
    for (int kk = 0; kk < 16; ++kk) {
      int k0 = kk * 32;
      v16bf a = load_a_frag(arow + k0 + half * 16, arow + k0 + half * 16 + 8);
      const bf16_t* b0 = socWb + (size_t)( 0 + m16) * POOLW + k0 + half * 16;
      const bf16_t* b1 = socWb + (size_t)(16 + m16) * POOLW + k0 + half * 16;
      const bf16_t* b2 = socWb + (size_t)(32 + m16) * POOLW + k0 + half * 16;
      const bf16_t* b3 = socWb + (size_t)(48 + m16) * POOLW + k0 + half * 16;
      acc0 = wmma_bf16(a, load_b_frag(b0), acc0);
      acc1 = wmma_bf16(a, load_b_frag(b1), acc1);
      acc2 = wmma_bf16(a, load_b_frag(b2), acc2);
      acc3 = wmma_bf16(a, load_b_frag(b3), acc3);
    }
    v8f accs[4] = {acc0, acc1, acc2, acc3};
#pragma unroll
    for (int nt = 0; nt < 4; ++nt) {
      int col = nt * 16 + m16;
      float bias = socb[col];
#pragma unroll
      for (int e = 0; e < 8; ++e) {
        int M = e + 8 * half;
        float v = accs[nt][e] + bias;
        xrow[M * XW + 64 + col] = (bf16_t)fmaxf(v, 0.0f);
      }
    }
  }
  __syncthreads();

  // ---- z = xh @ Wz^T ; gates ; masked h/c update.  K=256, 8 column bands ----
  for (int cb = 0; cb < 8; ++cb) {
    v8f ai = vzero8(), af = vzero8(), ag = vzero8(), ao = vzero8();
    const bf16_t* albase = &xrow[m16 * XW];
    int ncol = cb * 16 + m16;
#pragma unroll
    for (int kk = 0; kk < 8; ++kk) {
      int k0 = kk * 32;
      v16bf a = load_a_frag(albase + k0 + half * 16, albase + k0 + half * 16 + 8);
      ai = wmma_bf16(a, load_b_frag(Wz + (size_t)(  0 + ncol) * XW + k0 + half * 16), ai);
      af = wmma_bf16(a, load_b_frag(Wz + (size_t)(128 + ncol) * XW + k0 + half * 16), af);
      ag = wmma_bf16(a, load_b_frag(Wz + (size_t)(256 + ncol) * XW + k0 + half * 16), ag);
      ao = wmma_bf16(a, load_b_frag(Wz + (size_t)(384 + ncol) * XW + k0 + half * 16), ao);
    }
    int col = cb * 16 + m16;
    float bi = bih[      col] + bhh[      col];
    float bf = bih[128 + col] + bhh[128 + col];
    float bg = bih[256 + col] + bhh[256 + col];
    float bo = bih[384 + col] + bhh[384 + col];
#pragma unroll
    for (int e = 0; e < 8; ++e) {
      int row = rowBase + e + 8 * half;
      float zi = ai[e] + bi, zf = af[e] + bf, zg = ag[e] + bg, zo = ao[e] + bo;
      float cold = ct[row * HID + col];
      float hold = ht[row * HID + col];
      float cn = sigm(zf) * cold + sigm(zi) * tanhf(zg);
      float hn = sigm(zo) * tanhf(cn);
      bool act = (mask_base[row * mask_stride] == 1);
      ht[row * HID + col] = act ? hn : hold;
      ct[row * HID + col] = act ? cn : cold;
    }
  }
}

// =====================================================================
// prediction head: out = ht @ predW^T + predb ; rel = out[:,:2] ; pos += rel
// writes pred_out_all / pred_traj_all (masked). 1 thread per agent.
// =====================================================================
__global__ void sl_predout(const float* __restrict__ ht,
                           const float* __restrict__ predW, const float* __restrict__ predb,
                           const int* __restrict__ mask,
                           float* __restrict__ posbuf, float* __restrict__ relbuf,
                           float* __restrict__ out, int t) {
  int i = blockIdx.x * blockDim.x + threadIdx.x;
  if (i >= NAG) return;
  const float* h = ht + (size_t)i * HID;
  float o[5];
#pragma unroll
  for (int j = 0; j < 5; ++j) {
    float s = predb[j];
    const float* wr = predW + j * HID;
    for (int k = 0; k < HID; ++k) s += h[k] * wr[k];
    o[j] = s;
  }
  relbuf[i * 2 + 0] = o[0];
  relbuf[i * 2 + 1] = o[1];
  float px = posbuf[i * 2 + 0] + o[0];
  float py = posbuf[i * 2 + 1] + o[1];
  posbuf[i * 2 + 0] = px;
  posbuf[i * 2 + 1] = py;
  bool act = (mask[i * TOBS + (TOBS - 1)] == 1);
  float* po = out + (size_t)i * PREDLEN * 5 + (size_t)t * 5;
#pragma unroll
  for (int j = 0; j < 5; ++j) po[j] = act ? o[j] : 0.0f;
  float* pt = out + (size_t)NAG * PREDLEN * 5 + (size_t)i * PREDLEN * 2 + (size_t)t * 2;
  pt[0] = act ? px : 0.0f;
  pt[1] = act ? py : 0.0f;
}

__global__ void sl_initpos(const float* __restrict__ past_traj, float* __restrict__ posbuf) {
  int i = blockIdx.x * blockDim.x + threadIdx.x;   // 2048
  if (i < NAG * 2) {
    int row = i >> 1, d = i & 1;
    posbuf[i] = past_traj[row * TOBS * 2 + (TOBS - 1) * 2 + d];
  }
}

// =====================================================================
// host launcher
// =====================================================================
extern "C" void kernel_launch(void* const* d_in, const int* in_sizes, int n_in,
                              void* d_out, int out_size, void* d_ws, size_t ws_size,
                              hipStream_t stream) {
  (void)in_sizes; (void)n_in; (void)out_size; (void)ws_size;
  const float* past_traj = (const float*)d_in[0];
  const float* past_rel  = (const float*)d_in[1];
  const int*   mask      = (const int*)d_in[2];
  /* d_in[3] is_predictable: unused by reference */
  const int*   scene     = (const int*)d_in[4];
  const float* pos_W     = (const float*)d_in[5];
  const float* pos_b     = (const float*)d_in[6];
  const float* soc_W     = (const float*)d_in[7];
  const float* soc_b     = (const float*)d_in[8];
  const float* Wih       = (const float*)d_in[9];
  const float* Whh       = (const float*)d_in[10];
  const float* bih       = (const float*)d_in[11];
  const float* bhh       = (const float*)d_in[12];
  const float* h0        = (const float*)d_in[13];
  const float* c0        = (const float*)d_in[14];
  const float* pred_W    = (const float*)d_in[15];
  const float* pred_b    = (const float*)d_in[16];
  float* out = (float*)d_out;

  char* ws = (char*)d_ws;
  size_t off = 0;
  bf16_t* Wz     = (bf16_t*)(ws + off); off += (size_t)ZW * XW * 2;        // 256 KB
  bf16_t* socWb  = (bf16_t*)(ws + off); off += (size_t)EMB * POOLW * 2;    //  64 KB
  float*  ht     = (float*) (ws + off); off += (size_t)NAG * HID * 4;      // 512 KB
  float*  ct     = (float*) (ws + off); off += (size_t)NAG * HID * 4;      // 512 KB
  bf16_t* pooled = (bf16_t*)(ws + off); off += (size_t)NAG * POOLW * 2;    //   1 MB
  float*  posbuf = (float*) (ws + off); off += (size_t)NAG * 2 * 4;
  float*  relbuf = (float*) (ws + off); off += (size_t)NAG * 2 * 4;

  sl_prep<<<512, 256, 0, stream>>>(Wih, Whh, soc_W, h0, c0, Wz, socWb, ht, ct);

  // ---- observation phase ----
  for (int t = 0; t < TOBS; ++t) {
    sl_pool<<<NAG / SCN, 256, 0, stream>>>(ht, past_traj + t * 2, TOBS * 2,
                                           mask + t, TOBS, scene, pooled);
    sl_step<<<16, 128, 0, stream>>>(past_rel + t * 2, TOBS * 2, mask + t, TOBS,
                                    pooled, Wz, socWb, pos_W, pos_b, soc_b,
                                    bih, bhh, ht, ct);
  }

  // ---- prediction phase ----
  sl_initpos<<<8, 256, 0, stream>>>(past_traj, posbuf);
  for (int t = 0; t < PREDLEN; ++t) {
    sl_predout<<<NAG / 256, 256, 0, stream>>>(ht, pred_W, pred_b, mask,
                                              posbuf, relbuf, out, t);
    sl_pool<<<NAG / SCN, 256, 0, stream>>>(ht, posbuf, 2,
                                           mask + (TOBS - 1), TOBS, scene, pooled);
    sl_step<<<16, 128, 0, stream>>>(relbuf, 2, mask + (TOBS - 1), TOBS,
                                    pooled, Wz, socWb, pos_W, pos_b, soc_b,
                                    bih, bhh, ht, ct);
  }
}
